// NeLDA_83940840833389
// MI455X (gfx1250) — compile-verified, matrix operand order checked by reading, not run on hardware
//
#include <hip/hip_runtime.h>
#include <hip/hip_bf16.h>
#include <math.h>

// ---------------- problem constants ----------------
#define NB   1024      // batch
#define NV   50000     // vocab (multiple of 16)
#define NH   100       // hidden
#define NHP  112       // hidden padded to 7*16
#define NK   200       // topics
#define NT   32768     // tokens
#define BN_EPS 1e-5f

// GEMM1: K over V. chunks of 32: 1562 full + 1 tail(16 valid)
#define G1_KCHUNKS ((NV + 31) / 32)            // 1563
#define G1_SLICES  96                          // K-split slices (waves)
#define G1_CPS     ((G1_KCHUNKS + G1_SLICES - 1) / G1_SLICES)  // 17
// GEMM2: K over NK. chunks: 6 full + 1 tail(8 valid)
#define G2_KC      ((NK + 31) / 32)            // 7
#define G2_NITER   8                           // col-groups per wave
#define G2_COLS_PER_BLOCK (8 * G2_NITER * 16)  // 1024
#define G2_NGRP    ((NV + G2_COLS_PER_BLOCK - 1) / G2_COLS_PER_BLOCK)  // 49

typedef __attribute__((ext_vector_type(16))) __bf16 v16bf;
typedef __attribute__((ext_vector_type(8)))  float  v8f;
typedef __attribute__((ext_vector_type(4)))  float  f4;

// ----------------------------------------------------------------------------
// A fragment (16x32 bf16): element j <-> k = kBase + (j<8 ? g*8+j : 16+g*8+j-8)
// => per lane: two contiguous 8-float runs at kBase+g*8 and kBase+16+g*8.
// full==true : both runs valid.  full==false: only run1 valid (tail chunks:
// V%32==16 covers run1 for both lane groups; K%32==8 covers run1 for g==0).
// ----------------------------------------------------------------------------
__device__ __forceinline__ v16bf load_a_frag_fast(const float* __restrict__ rowp,
                                                  int kBase, int g, bool full, bool run1ok) {
    const float* p = rowp + kBase + g * 8;
    f4 f0 = {}, f1 = {}, f2 = {}, f3 = {};
    if (run1ok) { f0 = *(const f4*)(p);      f1 = *(const f4*)(p + 4); }
    if (full)   { f2 = *(const f4*)(p + 16); f3 = *(const f4*)(p + 20); }
    v16bf a;
#pragma unroll
    for (int j = 0; j < 4; ++j) {
        a[j]      = (__bf16)f0[j];
        a[4 + j]  = (__bf16)f1[j];
        a[8 + j]  = (__bf16)f2[j];
        a[12 + j] = (__bf16)f3[j];
    }
    return a;
}

// ---------------- small helpers ----------------
__device__ __forceinline__ float softplus_f(float x) {
    return (x > 20.0f) ? x : log1pf(expf(x));
}

__global__ void zero_f32(float* p, int n) {
    int i = blockIdx.x * blockDim.x + threadIdx.x;
    if (i < n) p[i] = 0.0f;
}

// ----------------------------------------------------------------------------
// Pack a row-major weight W[K, ldw] into per-fragment bf16 layout:
//   out[((c*Ncols + n)*2 + g)*16 + j] = W[k*ldw + n], k = c*32 + koff(g,j)
// zero-filled for k >= K or n >= realN  => GEMM loops need no guards.
// one thread per (c, n, g), writes 16 contiguous bf16 (32 B).
// ----------------------------------------------------------------------------
__global__ void pack_b(const float* __restrict__ W, int ldw, int K, int realN,
                       int Ncols, int nchunks, __bf16* __restrict__ out) {
    int id = blockIdx.x * blockDim.x + threadIdx.x;
    const int total = nchunks * Ncols * 2;
    if (id >= total) return;
    const int g = id & 1;
    const int n = (id >> 1) % Ncols;
    const int c = (id >> 1) / Ncols;
    __bf16* o = out + (size_t)id * 16;
    const bool nok = (n < realN);
#pragma unroll
    for (int j = 0; j < 16; ++j) {
        const int k = c * 32 + ((j < 8) ? (g * 8 + j) : (16 + g * 8 + (j - 8)));
        const float f = (nok && k < K) ? W[(size_t)k * ldw + n] : 0.0f;
        o[j] = (__bf16)f;
    }
}

// Pack theta[B,NK] into A-fragment bf16 layout, zero-filled past K=200.
// out[((mt*G2_KC + c)*32 + lane)*16 + j]; one thread per (mt, c, lane).
__global__ void pack_a_theta(const float* __restrict__ theta, __bf16* __restrict__ out) {
    int id = blockIdx.x * blockDim.x + threadIdx.x;
    const int total = (NB / 16) * G2_KC * 32;
    if (id >= total) return;
    const int lane = id & 31;
    const int c = (id >> 5) % G2_KC;
    const int mt = (id >> 5) / G2_KC;
    const int m = mt * 16 + (lane & 15);
    const int g = (lane >> 4) & 1;
    __bf16* o = out + (size_t)id * 16;
#pragma unroll
    for (int j = 0; j < 16; ++j) {
        const int k = c * 32 + ((j < 8) ? (g * 8 + j) : (16 + g * 8 + (j - 8)));
        o[j] = (__bf16)((k < NK) ? theta[m * NK + k] : 0.0f);
    }
}

// ----------------------------------------------------------------------------
// GEMM1: h1 += bows[B,V] @ fc1_w[V,H]
// grid (64 M-tiles, G1_SLICES/8), 256 thr = 8 waves.
// Each wave owns (M-tile, K-slice) and ALL 7 N-tiles (7 accumulators), so
// bows is streamed exactly once device-wide. B frags from packed bf16 (L2).
// ----------------------------------------------------------------------------
__global__ void gemm1_wmma(const float* __restrict__ bows,
                           const __bf16* __restrict__ Bpack,
                           float* __restrict__ h1) {
    const int wave  = threadIdx.x >> 5;
    const int lane  = threadIdx.x & 31;
    const int g     = (lane >> 4) & 1;
    const int mBase = blockIdx.x * 16;
    const int slice = blockIdx.y * 8 + wave;
    const int c0 = slice * G1_CPS;
    const int c1 = min(G1_KCHUNKS, c0 + G1_CPS);

    const float* rowp = bows + (size_t)(mBase + (lane & 15)) * NV;

    v8f acc[7];
#pragma unroll
    for (int t = 0; t < 7; ++t) acc[t] = (v8f){};

    for (int c = c0; c < c1; ++c) {
        const int kBase = c * 32;
        const bool full = (kBase + 32 <= NV);      // tail: V%32==16 -> run1 ok for both g
        v16bf a = load_a_frag_fast(rowp, kBase, g, full, true);
        const __bf16* bp = Bpack + ((size_t)(c * NHP + (lane & 15)) * 2 + g) * 16;
#pragma unroll
        for (int t = 0; t < 7; ++t) {
            v16bf b = *(const v16bf*)(bp + (size_t)t * 16 * 2 * 16);  // n += 16 cols
            acc[t] = __builtin_amdgcn_wmma_f32_16x16x32_bf16(false, a, false, b,
                                                             (short)0, acc[t], false, false);
        }
    }
    // C layout: M = r + 8*g, N = lane%16
#pragma unroll
    for (int t = 0; t < 7; ++t) {
        const int n = t * 16 + (lane & 15);
        if (n < NH) {
#pragma unroll
            for (int r = 0; r < 8; ++r)
                atomicAdd(&h1[(mBase + r + 8 * g) * NH + n], acc[t][r]);
        }
    }
}

// h1 = softplus(h1 + fc1_b)
__global__ void act1(float* __restrict__ h1, const float* __restrict__ fc1_b) {
    int i = blockIdx.x * blockDim.x + threadIdx.x;
    if (i < NB * NH) h1[i] = softplus_f(h1[i] + fc1_b[i % NH]);
}

// h2 = softplus(h1 @ fc2_w + fc2_b) : tiny 1024x100x100
__global__ void fc2_kernel(const float* __restrict__ h1,
                           const float* __restrict__ fc2_w,
                           const float* __restrict__ fc2_b,
                           float* __restrict__ h2) {
    int i = blockIdx.x * blockDim.x + threadIdx.x;
    if (i >= NB * NH) return;
    const int b = i / NH, j = i % NH;
    float s = fc2_b[j];
    for (int k = 0; k < NH; ++k) s += h1[b * NH + k] * fc2_w[k * NH + j];
    h2[i] = softplus_f(s);
}

// mu = h2@mu_w + mu_b ; sig = h2@sig_w + sig_b : 1024x200, K=100
__global__ void musig_kernel(const float* __restrict__ h2,
                             const float* __restrict__ mu_w, const float* __restrict__ mu_b,
                             const float* __restrict__ sig_w, const float* __restrict__ sig_b,
                             float* __restrict__ mu, float* __restrict__ sig) {
    int i = blockIdx.x * blockDim.x + threadIdx.x;
    if (i >= NB * NK) return;
    const int b = i / NK, k = i % NK;
    float sm = mu_b[k], ss = sig_b[k];
    for (int j = 0; j < NH; ++j) {
        const float h = h2[b * NH + j];
        sm += h * mu_w[j * NK + k];
        ss += h * sig_w[j * NK + k];
    }
    mu[i] = sm;
    sig[i] = ss;
}

// Batch-stat (mean, rstd) per feature of a [NB, C] matrix (coalesced columns).
__global__ void colstats(const float* __restrict__ x, int C,
                         float* __restrict__ mean, float* __restrict__ rstd) {
    int c = blockIdx.x * blockDim.x + threadIdx.x;
    if (c >= C) return;
    float s = 0.0f, s2 = 0.0f;
    for (int r = 0; r < NB; ++r) {
        const float v = x[(size_t)r * C + c];
        s += v; s2 += v * v;
    }
    const float m = s * (1.0f / NB);
    float var = fmaxf(s2 * (1.0f / NB) - m * m, 0.0f);
    mean[c] = m;
    rstd[c] = rsqrtf(var + BN_EPS);
}

// theta = exp(bn(mu) + exp(0.5*bn(sig)) * eps)
__global__ void theta_kernel(const float* __restrict__ mu, const float* __restrict__ sig,
                             const float* __restrict__ eps,
                             const float* __restrict__ m_mu, const float* __restrict__ r_mu,
                             const float* __restrict__ g_mu, const float* __restrict__ b_mu,
                             const float* __restrict__ m_sg, const float* __restrict__ r_sg,
                             const float* __restrict__ g_sg, const float* __restrict__ b_sg,
                             float* __restrict__ theta) {
    int i = blockIdx.x * blockDim.x + threadIdx.x;
    if (i >= NB * NK) return;
    const int k = i % NK;
    const float pm = g_mu[k] * (mu[i] - m_mu[k]) * r_mu[k] + b_mu[k];
    const float ps = expf(0.5f * (g_sg[k] * (sig[i] - m_sg[k]) * r_sg[k] + b_sg[k]));
    theta[i] = expf(pm + ps * eps[i]);
}

// ----------------------------------------------------------------------------
// GEMM2: logits = theta[B,K] @ beta_w[K,V] + beta_b
// grid (G2_NGRP, 64), 256 thr = 8 waves. Wave preloads its 7 A frags (packed
// bf16 theta), then sweeps G2_NITER 16-col groups: pure b128 loads + wmma.
// ----------------------------------------------------------------------------
__global__ void gemm2_wmma(const __bf16* __restrict__ Apack,
                           const __bf16* __restrict__ Bpack,
                           const float* __restrict__ beta_b,
                           float* __restrict__ logits) {
    const int wave  = threadIdx.x >> 5;
    const int lane  = threadIdx.x & 31;
    const int g     = (lane >> 4) & 1;
    const int mt    = blockIdx.y;
    const int mBase = mt * 16;

    v16bf af[G2_KC];
#pragma unroll
    for (int c = 0; c < G2_KC; ++c)
        af[c] = *(const v16bf*)(Apack + ((size_t)(mt * G2_KC + c) * 32 + lane) * 16);

    const int nStart = blockIdx.x * G2_COLS_PER_BLOCK + wave * (G2_NITER * 16);
#pragma unroll
    for (int it = 0; it < G2_NITER; ++it) {
        const int n16 = nStart + it * 16;
        if (n16 >= NV) break;                    // NV % 16 == 0: whole tiles only
        const int n = n16 + (lane & 15);
        if (it + 1 < G2_NITER && n16 + 16 < NV)  // hint next col-group of B
            __builtin_prefetch(Bpack + ((size_t)(n16 + 16 + (lane & 15)) * 2 + g) * 16, 0, 1);
        v8f acc = {};
#pragma unroll
        for (int c = 0; c < G2_KC; ++c) {
            v16bf b = *(const v16bf*)(Bpack + ((size_t)(c * NV + n) * 2 + g) * 16);
            acc = __builtin_amdgcn_wmma_f32_16x16x32_bf16(false, af[c], false, b,
                                                          (short)0, acc, false, false);
        }
        const float bb = beta_b[n];
#pragma unroll
        for (int r = 0; r < 8; ++r)
            logits[(size_t)(mBase + r + 8 * g) * NV + n] = acc[r] + bb;
    }
}

// normalize logits in place with the [V]-feature batchnorm
__global__ void bn_inplace(float* __restrict__ logits,
                           const float* __restrict__ cmean, const float* __restrict__ crstd,
                           const float* __restrict__ g, const float* __restrict__ b) {
    long i = (long)blockIdx.x * blockDim.x + threadIdx.x;
    if (i >= (long)NB * NV) return;
    const int c = (int)(i % NV);
    logits[i] = g[c] * (logits[i] - cmean[c]) * crstd[c] + b[c];
}

// per-row online logsumexp over V (one block per row)
__global__ void row_lse(const float* __restrict__ y, float* __restrict__ lse) {
    __shared__ float sm[256], ss[256];
    const int row = blockIdx.x, tid = threadIdx.x;
    float m = -INFINITY, s = 0.0f;
    for (int c = tid; c < NV; c += 256) {
        const float v = y[(size_t)row * NV + c];
        if (v > m) { s = s * expf(m - v) + 1.0f; m = v; }
        else       { s += expf(v - m); }
    }
    sm[tid] = m; ss[tid] = s;
    __syncthreads();
    for (int off = 128; off > 0; off >>= 1) {
        if (tid < off) {
            const float m2 = sm[tid + off], s2 = ss[tid + off];
            const float M = fmaxf(sm[tid], m2);
            ss[tid] = ss[tid] * expf(sm[tid] - M) + s2 * expf(m2 - M);
            sm[tid] = M;
        }
        __syncthreads();
    }
    if (tid == 0) lse[row] = sm[0] + logf(ss[0]);
}

// loglik[b] = sum_t y[b, tok[t]] - T * lse[b]
__global__ void gather_loglik(const float* __restrict__ y, const int* __restrict__ tok,
                              const float* __restrict__ lse, float* __restrict__ out) {
    __shared__ float sacc[256];
    const int row = blockIdx.x, tid = threadIdx.x;
    float acc = 0.0f;
    for (int t = tid; t < NT; t += 256) acc += y[(size_t)row * NV + tok[t]];
    sacc[tid] = acc;
    __syncthreads();
    for (int off = 128; off > 0; off >>= 1) {
        if (tid < off) sacc[tid] += sacc[tid + off];
        __syncthreads();
    }
    if (tid == 0) out[row] = sacc[0] - (float)NT * lse[row];
}

// ---------------- host driver ----------------
extern "C" void kernel_launch(void* const* d_in, const int* in_sizes, int n_in,
                              void* d_out, int out_size, void* d_ws, size_t ws_size,
                              hipStream_t stream) {
    const float* bows    = (const float*)d_in[0];
    const float* eps     = (const float*)d_in[1];
    const int*   tokens  = (const int*)  d_in[2];
    const float* fc1_w   = (const float*)d_in[3];
    const float* fc1_b   = (const float*)d_in[4];
    const float* fc2_w   = (const float*)d_in[5];
    const float* fc2_b   = (const float*)d_in[6];
    const float* mu_w    = (const float*)d_in[7];
    const float* mu_b    = (const float*)d_in[8];
    const float* bn_mu_g = (const float*)d_in[9];
    const float* bn_mu_b = (const float*)d_in[10];
    const float* sig_w   = (const float*)d_in[11];
    const float* sig_b   = (const float*)d_in[12];
    const float* bn_sg_g = (const float*)d_in[13];
    const float* bn_sg_b = (const float*)d_in[14];
    const float* beta_w  = (const float*)d_in[15];
    const float* beta_b  = (const float*)d_in[16];
    const float* bn_bt_g = (const float*)d_in[17];
    const float* bn_bt_b = (const float*)d_in[18];
    float* out = (float*)d_out;

    // ---- workspace carving ----
    char* base = (char*)d_ws;
    size_t off = 0;
    auto allocF = [&](size_t n) { float* p = (float*)(base + off); off += n * sizeof(float); return p; };
    float* h1     = allocF(NB * NH);
    float* h2     = allocF(NB * NH);
    float* mu     = allocF(NB * NK);
    float* sig    = allocF(NB * NK);
    float* theta  = allocF(NB * NK);
    float* m_mu   = allocF(NK);
    float* r_mu   = allocF(NK);
    float* m_sg   = allocF(NK);
    float* r_sg   = allocF(NK);
    float* lse    = allocF(NB);
    float* cmean  = allocF(NV);
    float* crstd  = allocF(NV);
    float* logits = allocF((size_t)NB * NV);          // 204.8 MB
    off = (off + 31) & ~(size_t)31;                    // 32B align bf16 buffers
    __bf16* Bpack1 = (__bf16*)(base + off); off += (size_t)G1_KCHUNKS * NHP * 32 * 2;   // 11.2 MB
    off = (off + 31) & ~(size_t)31;
    __bf16* Bpack2 = (__bf16*)(base + off); off += (size_t)G2_KC * NV * 32 * 2;         // 22.4 MB
    off = (off + 31) & ~(size_t)31;
    __bf16* Apack  = (__bf16*)(base + off); off += (size_t)(NB / 16) * G2_KC * 32 * 16 * 2;

    // 0) pack weights into bf16 WMMA-fragment layout (zero-filled padding)
    {
        int t1 = G1_KCHUNKS * NHP * 2;
        pack_b<<<(t1 + 255) / 256, 256, 0, stream>>>(fc1_w, NH, NV, NH, NHP, G1_KCHUNKS, Bpack1);
        int t2 = G2_KC * NV * 2;
        pack_b<<<(t2 + 255) / 256, 256, 0, stream>>>(beta_w, NV, NK, NV, NV, G2_KC, Bpack2);
    }

    // 1) zero GEMM1 accumulator
    zero_f32<<<(NB * NH + 255) / 256, 256, 0, stream>>>(h1, NB * NH);

    // 2) GEMM1 (WMMA bf16; wave owns M-tile x K-slice x full H; atomic fp32)
    gemm1_wmma<<<dim3(NB / 16, G1_SLICES / 8), 256, 0, stream>>>(bows, Bpack1, h1);

    // 3) bias + softplus
    act1<<<(NB * NH + 255) / 256, 256, 0, stream>>>(h1, fc1_b);

    // 4) fc2 + softplus
    fc2_kernel<<<(NB * NH + 255) / 256, 256, 0, stream>>>(h1, fc2_w, fc2_b, h2);

    // 5) mu / sigma heads
    musig_kernel<<<(NB * NK + 255) / 256, 256, 0, stream>>>(h2, mu_w, mu_b, sig_w, sig_b, mu, sig);

    // 6) batch stats for mu and sigma heads
    colstats<<<(NK + 255) / 256, 256, 0, stream>>>(mu, NK, m_mu, r_mu);
    colstats<<<(NK + 255) / 256, 256, 0, stream>>>(sig, NK, m_sg, r_sg);

    // 7) theta
    theta_kernel<<<(NB * NK + 255) / 256, 256, 0, stream>>>(
        mu, sig, eps, m_mu, r_mu, bn_mu_g, bn_mu_b, m_sg, r_sg, bn_sg_g, bn_sg_b, theta);

    // 7b) pack theta into A-fragment bf16 layout
    {
        int ta = (NB / 16) * G2_KC * 32;
        pack_a_theta<<<(ta + 255) / 256, 256, 0, stream>>>(theta, Apack);
    }

    // 8) GEMM2 (WMMA bf16) -> raw logits + beta_b
    gemm2_wmma<<<dim3(G2_NGRP, NB / 16), 256, 0, stream>>>(Apack, Bpack2, beta_b, logits);

    // 9) batch stats per vocab feature
    colstats<<<(NV + 255) / 256, 256, 0, stream>>>(logits, NV, cmean, crstd);

    // 10) batchnorm in place
    {
        long n = (long)NB * NV;
        bn_inplace<<<(unsigned)((n + 255) / 256), 256, 0, stream>>>(logits, cmean, crstd, bn_bt_g, bn_bt_b);
    }

    // 11) per-row online logsumexp
    row_lse<<<NB, 256, 0, stream>>>(logits, lse);

    // 12) token gather + sum -> loglik
    gather_loglik<<<NB, 256, 0, stream>>>(logits, tokens, lse, out);

    (void)in_sizes; (void)n_in; (void)out_size; (void)ws_size;
}
